// HelixMambaBackbone_89979564852032
// MI455X (gfx1250) — matrix operand order repaired and checked by reading, not compiled, not run on hardware
//
#include <hip/hip_runtime.h>
#include <hip/hip_bf16.h>
#include <math.h>

#define H 256
#define BATCH 32
#define MAXN 64
#define NNODES 2048
#define MPROT 512
#define DIN 512
#define DCONV 4

typedef unsigned short ushort_t;
typedef __attribute__((ext_vector_type(16))) __bf16 bf16x16;
typedef __attribute__((ext_vector_type(8)))  float  floatx8;

__device__ __forceinline__ ushort_t f2bf(float f) {
  unsigned int u = __float_as_uint(f);
  unsigned int r = (u + 0x7FFFu + ((u >> 16) & 1u)) >> 16;  // RNE
  return (ushort_t)r;
}
__device__ __forceinline__ float sigmoidf_(float x) { return 1.0f / (1.0f + __expf(-x)); }
__device__ __forceinline__ float siluf_(float x)    { return x * sigmoidf_(x); }
__device__ __forceinline__ float softplusf_(float x){ return fmaxf(x, 0.0f) + log1pf(__expf(-fabsf(x))); }

// A operand, fragment-major pack: element (m, k) of a [Mrows, K] bf16 matrix lives at
//   Ap[(k/32)*Mrows*32 + m*32 + (k%32)]
// so a wave's 16x32 A-fragment is one contiguous 1KB block (16 rows x 64B).
__device__ __forceinline__ size_t apack_idx(int m, int k, int Mrows) {
  return (size_t)(k >> 5) * Mrows * 32 + (size_t)m * 32 + (k & 31);
}

// ---------------------------------------------------------------------------
// Weight prep: W[Nrows, ldW] f32 (slice at col0, Kc cols) -> B fragment-major pack:
//   Bp[((k/32)*(Nrows/16) + n/16)*512 + (n%16)*32 + (k%32)]
// Lane L of a WMMA B-frag (n = L%16, k = 8h+s / 16+8h+s) loads base + (L%16)*64B + h*16B.
// ---------------------------------------------------------------------------
__global__ void k_wpackB(const float* __restrict__ W, ushort_t* __restrict__ Bp,
                         int Nrows, int Kc, int ldW, int col0) {
  int idx = blockIdx.x * 256 + threadIdx.x;
  if (idx >= Nrows * Kc) return;
  int k = idx / Nrows, n = idx - k * Nrows;
  size_t o = ((size_t)(k >> 5) * (Nrows >> 4) + (n >> 4)) * 512 + (size_t)(n & 15) * 32 + (k & 31);
  Bp[o] = f2bf(W[(size_t)n * ldW + col0 + k]);
}

// GVP effective bias: c[o] = bs[o] + 1e-6 * sum_{j=256..271} Ws[o, j]   (v == 0 identity)
__global__ void k_gvpc(const float* __restrict__ Ws, const float* __restrict__ bs,
                       float* __restrict__ c) {
  int o = threadIdx.x;
  float acc = bs[o];
#pragma unroll
  for (int j = 0; j < 16; ++j) acc += 1e-6f * Ws[(size_t)o * 272 + 256 + j];
  c[o] = acc;
}

// ---------------------------------------------------------------------------
// Time embedding MLP: one block per batch element.
// ---------------------------------------------------------------------------
__global__ __launch_bounds__(256)
void k_temb(const float* __restrict__ t, const float* __restrict__ Wt1,
            const float* __restrict__ bt1, const float* __restrict__ Wt2,
            const float* __restrict__ bt2, float* __restrict__ te_out) {
  __shared__ float te[H];
  __shared__ float h1[2 * H];
  int b = blockIdx.x, tid = threadIdx.x;
  if (tid < 128) {
    float f = __expf((float)tid * (-logf(10000.0f) / 127.0f));
    float v = t[b] * f;
    te[tid] = sinf(v);
    te[tid + 128] = cosf(v);
  }
  __syncthreads();
  for (int o = tid; o < 2 * H; o += 256) {
    float acc = bt1[o];
    const float* w = Wt1 + (size_t)o * H;
    for (int k = 0; k < H; ++k) acc += te[k] * w[k];
    h1[o] = siluf_(acc);
  }
  __syncthreads();
  {
    float acc = bt2[tid];
    const float* w = Wt2 + (size_t)tid * 512;
    for (int k = 0; k < 512; ++k) acc += h1[k] * w[k];
    te_out[(size_t)b * H + tid] = acc;
  }
}

// prox[n] = exp(-min_a ||pos_L[n]-pos_P[a]||)
__global__ __launch_bounds__(256)
void k_prox(const float* __restrict__ posL, const float* __restrict__ posP,
            float* __restrict__ prox) {
  __shared__ float red[256];
  int n = blockIdx.x, tid = threadIdx.x;
  float px = posL[n * 3], py = posL[n * 3 + 1], pz = posL[n * 3 + 2];
  float mn = 3.4e38f;
  for (int a = tid; a < MPROT; a += 256) {
    float dx = px - posP[a * 3], dy = py - posP[a * 3 + 1], dz = pz - posP[a * 3 + 2];
    mn = fminf(mn, dx * dx + dy * dy + dz * dz);
  }
  red[tid] = mn;
  __syncthreads();
  for (int s = 128; s > 0; s >>= 1) {
    if (tid < s) red[tid] = fminf(red[tid], red[tid + s]);
    __syncthreads();
  }
  if (tid == 0) prox[n] = __expf(-sqrtf(fmaxf(red[0], 1e-12f)));
}

// s[n,h] = (x_L[n]@W_emb^T + b_emb + te[batch[n]]) * (1 + prox[n]); bf16 copy A-packed
__global__ __launch_bounds__(256)
void k_embed(const float* __restrict__ xL, const float* __restrict__ Wemb,
             const float* __restrict__ bemb, const float* __restrict__ te,
             const int* __restrict__ batch, const float* __restrict__ prox,
             float* __restrict__ s, ushort_t* __restrict__ sbf) {
  int n = blockIdx.x, h = threadIdx.x;
  float acc = bemb[h];
  const float* xr = xL + (size_t)n * 9;
  const float* wr = Wemb + (size_t)h * 9;
#pragma unroll
  for (int j = 0; j < 9; ++j) acc += xr[j] * wr[j];
  acc += te[(size_t)batch[n] * H + h];
  float v = acc * (1.0f + prox[n]);
  s[(size_t)n * H + h] = v;
  sbf[apack_idx(n, h, NNODES)] = f2bf(v);
}

// ---------------------------------------------------------------------------
// WMMA GEMM: C[M,N] = act(A @ W^T + bias) with fragment-major packed operands.
//   A:  apack layout of [M,K] bf16 activations
//   Bp: k_wpackB layout of W (so Bp holds W^T tiles)
// Wave tile 16x64 (4 accumulators), block = 8 waves -> 128x64 block tile.
// Every operand load is a contiguous 1KB/wave (32 lanes x 32B). No LDS needed.
// act: 0=none 1=softplus 2=sigmoid 3=tanh.  Cbf (optional) written A-packed.
// ---------------------------------------------------------------------------
__global__ __launch_bounds__(256)
void k_gemm_wmma(const ushort_t* __restrict__ A, const ushort_t* __restrict__ Bp,
                 const float* __restrict__ bias, float* __restrict__ C,
                 ushort_t* __restrict__ Cbf, int M, int N, int K, int act) {
  const int lane = threadIdx.x & 31;
  const int wave = threadIdx.x >> 5;
  const int m0 = blockIdx.y * 128 + wave * 16;
  const int n0 = blockIdx.x * 64;
  const int mrow = lane & 15;
  const int hf = lane >> 4;

  floatx8 acc[4] = {};

  const ushort_t* abase = A + (size_t)(m0 + mrow) * 32 + hf * 8;  // k-block 0
  const size_t astep = (size_t)M * 32;
  const ushort_t* bbase = Bp + (size_t)(n0 >> 4) * 512 + (size_t)(lane & 15) * 32 + hf * 8;  // tile n0/16
  const size_t bstep = (size_t)(N >> 4) * 512;

  for (int kb = 0; kb < K; kb += 32) {
    union { bf16x16 v; uint4 q[2]; } af;
    af.q[0] = *(const uint4*)(abase);        // K = kb + hf*8 .. +8
    af.q[1] = *(const uint4*)(abase + 16);   // K = kb+16+hf*8 .. +8
    if (kb + 32 < K) {                       // gfx1250 global_prefetch_b8 for next k-step
      __builtin_prefetch(abase + astep, 0, 3);
      __builtin_prefetch(bbase + bstep, 0, 3);
    }
#pragma unroll
    for (int t = 0; t < 4; ++t) {
      union { bf16x16 v; uint4 q[2]; } bfr;
      bfr.q[0] = *(const uint4*)(bbase + (size_t)t * 512);
      bfr.q[1] = *(const uint4*)(bbase + (size_t)t * 512 + 16);
      acc[t] = __builtin_amdgcn_wmma_f32_16x16x32_bf16(
          false, af.v, false, bfr.v, (short)0, acc[t], false, false);
    }
    abase += astep;
    bbase += bstep;
  }

#pragma unroll
  for (int t = 0; t < 4; ++t) {
#pragma unroll
    for (int r = 0; r < 8; ++r) {
      int m = m0 + r + 8 * hf;
      int n = n0 + t * 16 + (lane & 15);
      float v = acc[t][r];
      if (bias) v += bias[n];
      if (act == 1) v = softplusf_(v);
      else if (act == 2) v = sigmoidf_(v);
      else if (act == 3) v = tanhf(v);
      C[(size_t)m * N + n] = v;
      if (Cbf) Cbf[apack_idx(m, n, M)] = f2bf(v);
    }
  }
}

// ---------------------------------------------------------------------------
// Causal depthwise conv1d + SiLU; also silu(z).  block = 512 threads (one per d).
// ---------------------------------------------------------------------------
__global__ __launch_bounds__(512)
void k_conv(const float* __restrict__ xz, const float* __restrict__ convw,
            const float* __restrict__ convb, float* __restrict__ xc,
            ushort_t* __restrict__ xcbf, float* __restrict__ siluz) {
  int n = blockIdx.x, d = threadIdx.x;
  int b = n >> 6, l = n & 63;
  float acc = convb[d];
#pragma unroll
  for (int k = 0; k < DCONV; ++k) {
    int l2 = l - (DCONV - 1) + k;
    if (l2 >= 0) acc += xz[(size_t)(b * 64 + l2) * 1024 + d] * convw[d * 4 + k];
  }
  float v = siluf_(acc);
  size_t idx = (size_t)n * DIN + d;
  xc[idx] = v;
  xcbf[apack_idx(n, d, NNODES)] = f2bf(v);
  siluz[idx] = siluf_(xz[(size_t)n * 1024 + DIN + d]);
}

// ---------------------------------------------------------------------------
// Selective-scan (log-space cumsum trick).  One thread per (b, d) channel.
// m = max_l(-S) == -S[63] since la < 0 (S strictly decreasing).
// ---------------------------------------------------------------------------
__global__ __launch_bounds__(512)
void k_scan(const float* __restrict__ dt, const float* __restrict__ xc,
            const float* __restrict__ Btb, const float* __restrict__ Ctb,
            const float* __restrict__ sz, const float* __restrict__ A_log,
            float* __restrict__ y, ushort_t* __restrict__ ybf) {
  int b = blockIdx.x, d = threadIdx.x;
  float Ad = -__expf(A_log[d]);
  float S = 0.0f;
  for (int l = 0; l < MAXN; ++l) {
    size_t idx = (size_t)(b * 64 + l) * DIN + d;
    S += fminf(fmaxf(Ad * dt[idx], -10.0f), -1e-4f);
  }
  float m = -S;
  S = 0.0f;
  float acc = 0.0f, uprev = 0.0f;
  for (int l = 0; l < MAXN; ++l) {
    size_t idx = (size_t)(b * 64 + l) * DIN + d;
    float dtv = dt[idx];
    S += fminf(fmaxf(Ad * dtv, -10.0f), -1e-4f);
    float u = xc[idx] * Btb[idx];
    float beta = dtv * 0.5f * (uprev + u);
    uprev = u;
    acc += beta * __expf(-S - m);
    float hval = __expf(S + m) * acc;
    float yv = hval * Ctb[idx] * sz[idx];
    y[idx] = yv;
    ybf[apack_idx(b * 64 + l, d, NNODES)] = f2bf(yv);
  }
}

// ---------------------------------------------------------------------------
// Pair epilogue: v_pred[b,i] = sum_j (Wp2·silu(P[j] + Q[i] + dist*w1c) + bp2)*(pos_j-pos_i)
// One block per (b,i); each wave handles j = wave, wave+8, ...; lane covers 8 channels.
// ---------------------------------------------------------------------------
__global__ __launch_bounds__(256)
void k_pair(const float* __restrict__ P, const float* __restrict__ Q,
            const float* __restrict__ posL, const float* __restrict__ Wp1,
            const float* __restrict__ Wp2, const float* __restrict__ bp2,
            float* __restrict__ out) {
  __shared__ float posS[MAXN * 3];
  __shared__ float wacc[8][3];
  int bi = blockIdx.x;
  int b = bi >> 6, i = bi & 63;
  int tid = threadIdx.x, lane = tid & 31, wave = tid >> 5;
  if (tid < MAXN * 3) posS[tid] = posL[(size_t)b * MAXN * 3 + tid];
  __syncthreads();

  float qreg[8], w1c[8], w2[8];
  const float* Qrow = Q + (size_t)bi * H;
#pragma unroll
  for (int c = 0; c < 8; ++c) {
    int k = lane + 32 * c;
    qreg[c] = Qrow[k];
    w1c[c] = Wp1[(size_t)k * 513 + 512];
    w2[c] = Wp2[k];
  }
  float pix = posS[i * 3], piy = posS[i * 3 + 1], piz = posS[i * 3 + 2];
  float b2 = bp2[0];
  float ax = 0.f, ay = 0.f, az = 0.f;
  const float* Pb = P + (size_t)b * MAXN * H;

  for (int j = wave; j < MAXN; j += 8) {
    float dx = posS[j * 3] - pix, dy = posS[j * 3 + 1] - piy, dz = posS[j * 3 + 2] - piz;
    float dist = sqrtf(fmaxf(dx * dx + dy * dy + dz * dz, 1e-12f));
    const float* Prow = Pb + (size_t)j * H;
    float part = 0.f;
#pragma unroll
    for (int c = 0; c < 8; ++c) {
      int k = lane + 32 * c;
      float pre = qreg[c] + Prow[k] + dist * w1c[c];
      part += siluf_(pre) * w2[c];
    }
#pragma unroll
    for (int off = 16; off >= 1; off >>= 1) part += __shfl_xor(part, off, 32);
    float coeff = part + b2;
    ax += coeff * dx; ay += coeff * dy; az += coeff * dz;
  }
  if (lane == 0) { wacc[wave][0] = ax; wacc[wave][1] = ay; wacc[wave][2] = az; }
  __syncthreads();
  if (tid == 0) {
    float sx = 0.f, sy = 0.f, sz2 = 0.f;
    for (int w = 0; w < 8; ++w) { sx += wacc[w][0]; sy += wacc[w][1]; sz2 += wacc[w][2]; }
    out[(size_t)bi * 3 + 0] = sx;
    out[(size_t)bi * 3 + 1] = sy;
    out[(size_t)bi * 3 + 2] = sz2;
  }
}

// ---------------------------------------------------------------------------
extern "C" void kernel_launch(void* const* d_in, const int* in_sizes, int n_in,
                              void* d_out, int out_size, void* d_ws, size_t ws_size,
                              hipStream_t stream) {
  (void)in_sizes; (void)n_in; (void)out_size; (void)ws_size;
  // inputs (setup_inputs order), then params as sorted pytree leaves
  const float* x_L   = (const float*)d_in[0];
  const float* pos_L = (const float*)d_in[1];
  const float* pos_P = (const float*)d_in[3];
  const float* tvec  = (const float*)d_in[4];
  const int*   batch = (const int*)d_in[5];
  const float* A_log = (const float*)d_in[6];
  const float* W_B   = (const float*)d_in[7];
  const float* W_C   = (const float*)d_in[8];
  const float* W_dt  = (const float*)d_in[9];
  const float* W_emb = (const float*)d_in[10];
  const float* W_in  = (const float*)d_in[11];
  const float* W_out = (const float*)d_in[12];
  const float* Wp1   = (const float*)d_in[13];
  const float* Wp2   = (const float*)d_in[14];
  const float* Wt1   = (const float*)d_in[15];
  const float* Wt2   = (const float*)d_in[16];
  const float* b_dt  = (const float*)d_in[17];
  const float* b_emb = (const float*)d_in[18];
  const float* bp1   = (const float*)d_in[19];
  const float* bp2   = (const float*)d_in[20];
  const float* bt1   = (const float*)d_in[21];
  const float* bt2   = (const float*)d_in[22];
  const float* conv_b = (const float*)d_in[23];
  const float* conv_w = (const float*)d_in[24];
  const float* Wsl[3]; const float* bsl[3];
  for (int l = 0; l < 3; ++l) {
    Wsl[l] = (const float*)d_in[26 + 5 * l];
    bsl[l] = (const float*)d_in[29 + 5 * l];
  }

  // workspace bump allocator
  char* base = (char*)d_ws;
  size_t off = 0;
  auto alloc = [&](size_t bytes) -> void* {
    void* p = base + off;
    off = (off + bytes + 255) & ~(size_t)255;
    return p;
  };
  // bf16 weight packs (fragment-major)
  ushort_t* Wsa_t[3];
  for (int l = 0; l < 3; ++l) Wsa_t[l] = (ushort_t*)alloc(256 * 256 * 2);
  float* cgvp[3];
  for (int l = 0; l < 3; ++l) cgvp[l] = (float*)alloc(256 * 4);
  ushort_t* W_in_t  = (ushort_t*)alloc((size_t)256 * 1024 * 2);
  ushort_t* W_dt_t  = (ushort_t*)alloc((size_t)512 * 512 * 2);
  ushort_t* W_B_t   = (ushort_t*)alloc((size_t)512 * 512 * 2);
  ushort_t* W_C_t   = (ushort_t*)alloc((size_t)512 * 512 * 2);
  ushort_t* W_out_t = (ushort_t*)alloc((size_t)512 * 256 * 2);
  ushort_t* Wp1a_t  = (ushort_t*)alloc((size_t)256 * 256 * 2);
  ushort_t* Wp1b_t  = (ushort_t*)alloc((size_t)256 * 256 * 2);
  // activations (bf16 copies stored A-packed)
  float*    te_out = (float*)alloc((size_t)BATCH * H * 4);
  float*    prox   = (float*)alloc((size_t)NNODES * 4);
  float*    sA     = (float*)alloc((size_t)NNODES * H * 4);
  ushort_t* sA_bf  = (ushort_t*)alloc((size_t)NNODES * H * 2);
  float*    sB     = (float*)alloc((size_t)NNODES * H * 4);
  ushort_t* sB_bf  = (ushort_t*)alloc((size_t)NNODES * H * 2);
  float*    xz     = (float*)alloc((size_t)NNODES * 1024 * 4);
  float*    xc     = (float*)alloc((size_t)NNODES * DIN * 4);
  ushort_t* xc_bf  = (ushort_t*)alloc((size_t)NNODES * DIN * 2);
  float*    sluz   = (float*)alloc((size_t)NNODES * DIN * 4);
  float*    dtb    = (float*)alloc((size_t)NNODES * DIN * 4);
  float*    Btb    = (float*)alloc((size_t)NNODES * DIN * 4);
  float*    Ctb    = (float*)alloc((size_t)NNODES * DIN * 4);
  float*    ybuf   = (float*)alloc((size_t)NNODES * DIN * 4);
  ushort_t* y_bf   = (ushort_t*)alloc((size_t)NNODES * DIN * 2);
  float*    Hp     = (float*)alloc((size_t)NNODES * H * 4);
  ushort_t* Hp_bf  = (ushort_t*)alloc((size_t)NNODES * H * 2);
  float*    Pbuf   = (float*)alloc((size_t)NNODES * H * 4);
  float*    Qbuf   = (float*)alloc((size_t)NNODES * H * 4);

  auto packB = [&](const float* W, ushort_t* Wt, int Nr, int Kc, int ldW, int col0) {
    int total = Nr * Kc;
    k_wpackB<<<dim3((total + 255) / 256), dim3(256), 0, stream>>>(W, Wt, Nr, Kc, ldW, col0);
  };
  auto gemm = [&](const ushort_t* A, const ushort_t* Bt, const float* bias,
                  float* C, ushort_t* Cbf, int M, int N, int K, int act) {
    k_gemm_wmma<<<dim3(N / 64, M / 128), dim3(256), 0, stream>>>(A, Bt, bias, C, Cbf, M, N, K, act);
  };

  // ---- weight prep ----
  for (int l = 0; l < 3; ++l) {
    packB(Wsl[l], Wsa_t[l], 256, 256, 272, 0);
    k_gvpc<<<dim3(1), dim3(256), 0, stream>>>(Wsl[l], bsl[l], cgvp[l]);
  }
  packB(W_in,  W_in_t,  1024, 256, 256, 0);
  packB(W_dt,  W_dt_t,  512, 512, 512, 0);
  packB(W_B,   W_B_t,   512, 512, 512, 0);
  packB(W_C,   W_C_t,   512, 512, 512, 0);
  packB(W_out, W_out_t, 256, 512, 512, 0);
  packB(Wp1,   Wp1a_t,  256, 256, 513, 0);
  packB(Wp1,   Wp1b_t,  256, 256, 513, 256);

  // ---- embedding ----
  k_temb<<<dim3(BATCH), dim3(256), 0, stream>>>(tvec, Wt1, bt1, Wt2, bt2, te_out);
  k_prox<<<dim3(NNODES), dim3(256), 0, stream>>>(pos_L, pos_P, prox);
  k_embed<<<dim3(NNODES), dim3(256), 0, stream>>>(x_L, W_emb, b_emb, te_out, batch, prox, sA, sA_bf);

  // ---- GVP stack (v == 0 => pure GEMMs) ----
  gemm(sA_bf, Wsa_t[0], cgvp[0], sB, sB_bf, NNODES, 256, 256, 0);
  gemm(sB_bf, Wsa_t[1], cgvp[1], sA, sA_bf, NNODES, 256, 256, 0);
  gemm(sA_bf, Wsa_t[2], cgvp[2], sB, sB_bf, NNODES, 256, 256, 0);

  // ---- mamba ----
  gemm(sB_bf, W_in_t, nullptr, xz, nullptr, NNODES, 1024, 256, 0);
  k_conv<<<dim3(NNODES), dim3(512), 0, stream>>>(xz, conv_w, conv_b, xc, xc_bf, sluz);
  gemm(xc_bf, W_dt_t, b_dt,    dtb, nullptr, NNODES, 512, 512, 1);  // softplus
  gemm(xc_bf, W_B_t,  nullptr, Btb, nullptr, NNODES, 512, 512, 2);  // sigmoid
  gemm(xc_bf, W_C_t,  nullptr, Ctb, nullptr, NNODES, 512, 512, 3);  // tanh
  k_scan<<<dim3(BATCH), dim3(512), 0, stream>>>(dtb, xc, Btb, Ctb, sluz, A_log, ybuf, y_bf);
  gemm(y_bf, W_out_t, nullptr, Hp, Hp_bf, NNODES, 256, 512, 0);

  // ---- pair MLP factorization ----
  gemm(Hp_bf, Wp1a_t, nullptr, Pbuf, nullptr, NNODES, 256, 256, 0);  // j-side
  gemm(Hp_bf, Wp1b_t, bp1,     Qbuf, nullptr, NNODES, 256, 256, 0);  // i-side (+bp1)
  k_pair<<<dim3(NNODES), dim3(256), 0, stream>>>(Pbuf, Qbuf, pos_L, Wp1, Wp2, bp2, (float*)d_out);
}
